// GraphAttentionEmbedder_32452772888848
// MI455X (gfx1250) — matrix operand order repaired and checked by reading, not compiled, not run on hardware
//
#include <hip/hip_runtime.h>
#include <math.h>

#define DEV __device__ __forceinline__

typedef __attribute__((ext_vector_type(16))) __bf16 v16bf;
typedef __attribute__((ext_vector_type(8)))  float  v8f;
typedef unsigned short u16;
typedef unsigned int   u32;

// ---- problem sizes (fixed by the reference) ----
#define NQ    100000
#define KNB   20
#define DN    128
#define DE    128
#define DQ    256
#define DKV   384
#define TILE  16
#define NWAVE 8

// ---- d_ws layout: bf16 weights (element offsets) ----
#define OWQ   0          // Wq      256x256 (row-major, out x in)
#define OWKT  65536      // Wk^T    384x256 (WkT[j][d] = Wk[d][j])
#define OWV   163840     // Wv      256x384
#define OWO   262144     // Wo      256x256
#define OW1   327680     // W1      128x384
#define OW2   376832     // W2      128x128
#define OQT   393216     // q_time const: cos(time_b[j]), 128
#define WS_ELEMS 393344  // * 2 bytes = 786,688 B of d_ws used

// ---- dynamic LDS layout (bytes); mo/mi/h1 alias the dead w~ region ----
#define OFF_WT   0        // w~ : 16 x 768 bf16 (24576; dead after attention)
#define OFF_MO   0        // mha out : 16 x 256 bf16 (8192)   [alias]
#define OFF_MI   8192     // m_in    : 16 x 384 bf16 (12288)  [alias]
#define OFF_H1   20480    // h1      : 16 x 128 bf16 (4096)   [alias]
#define OFF_QC   24576    // qc : 16 x 256 bf16 (8192)
#define OFF_QH   32768    // qh : 16 x 256 bf16 (8192)
#define OFF_CB   40960    // c  : 16 x 768 bf16 (24576)
#define OFF_IDX  65536    // 16 i32
#define OFF_NBR  65600    // 16x20 i32
#define OFF_EID  66880    // 16x20 i32
#define OFF_DT   68160    // 16x20 f32
#define OFF_MSK  69440    // 16x20 f32 (raw mask)
#define OFF_VAL  70720    // 16x20 f32 (mask | no_nb)
#define OFF_ROWM 72000    // 16 f32 (0 if no_nb)
#define SMEM_BYTES 72064  // -> 4 workgroups (32 waves) per WGP

DEV u16 f2bf(float f) {                       // f32 -> bf16, round-to-nearest-even
  u32 u = __builtin_bit_cast(u32, f);
  u32 r = (u + 0x7FFFu + ((u >> 16) & 1u)) >> 16;
  return (u16)r;
}
DEV float bf2f(u16 h) { u32 u = ((u32)h) << 16; return __builtin_bit_cast(float, u); }

DEV float wsum(float v) {
  #pragma unroll
  for (int o = 16; o > 0; o >>= 1) v += __shfl_xor(v, o);
  return v;
}
DEV float wmaxr(float v) {
  #pragma unroll
  for (int o = 16; o > 0; o >>= 1) v = fmaxf(v, __shfl_xor(v, o));
  return v;
}

struct Frag { union { v16bf v; u32 u[8]; }; };

DEV void loadA(Frag& f, const u16* arow, int ks, int aH) {
  #pragma unroll
  for (int r = 0; r < 8; ++r) {
    int ka = (r < 4) ? (aH + 2 * r) : (16 + aH + 2 * (r - 4));
    f.u[r] = *(const u32*)(arow + ks + ka);
  }
}
DEV void loadB(Frag& f, const u16* brow, int ks, int bH) {
  #pragma unroll
  for (int r = 0; r < 8; ++r) f.u[r] = *(const u32*)(brow + ks + ((r < 4) ? (bH + 2 * r) : (16 + bH + 2 * (r - 4))));
}

// D(16x16,f32) += A(16xK bf16, LDS row-major, stride lda) x B(Kx16 bf16, global)
// B(k,n) := Bg[(brow0+n)*ldb + bcol0 + k]  (two runs of 8 contiguous bf16 per lane -> b128s)
// K is a multiple of 64: unroll-by-2 with two B buffers so the next B tile is
// always in flight across the current WMMA (partial s_wait_loadcnt, not 0).
DEV v8f gemm16(const u16* __restrict__ A, int lda,
               const u16* __restrict__ B, int ldb, int brow0, int bcol0,
               int K, v8f acc) {
  const int lane = threadIdx.x & 31;
  const int m  = lane & 15;
  const int aH = (lane & 16) ? 8  : 0;   // A: high half-lanes hold K+8 / K+24
  const int bH = (lane & 16) ? 8  : 0;   // B: high half-lanes hold K+8 / K+24
  const u16* arow = A + m * lda;
  const u16* brow = B + (size_t)(brow0 + m) * ldb + bcol0;

  Frag a, b0, b1;
  loadB(b0, brow, 0, bH);
  for (int ks = 0; ks < K; ks += 64) {
    loadB(b1, brow, ks + 32, bH);
    loadA(a, arow, ks, aH);
    acc = __builtin_amdgcn_wmma_f32_16x16x32_bf16(false, a.v, false, b0.v,
                                                  (short)0, acc, false, false);
    if (ks + 64 < K) loadB(b0, brow, ks + 64, bH);
    loadA(a, arow, ks + 32, aH);
    acc = __builtin_amdgcn_wmma_f32_16x16x32_bf16(false, a.v, false, b1.v,
                                                  (short)0, acc, false, false);
  }
  return acc;
}

// ---- branch-free store epilogues ----
DEV void storeD_plain_bf(u16* out, int ldo, int col0, v8f d) {
  const int lane = threadIdx.x & 31;
  const int n = lane & 15, mb = (lane & 16) ? 8 : 0;
  #pragma unroll
  for (int r = 0; r < 8; ++r) out[(mb + r) * ldo + col0 + n] = f2bf(d[r]);
}
DEV void storeD_bias_bf(u16* out, int ldo, int col0, v8f d, const float* bias) {
  const int lane = threadIdx.x & 31;
  const int n = lane & 15, mb = (lane & 16) ? 8 : 0;
  float b = bias[col0 + n];
  #pragma unroll
  for (int r = 0; r < 8; ++r) out[(mb + r) * ldo + col0 + n] = f2bf(d[r] + b);
}
DEV void storeD_bias_row_bf(u16* out, int ldo, int col0, v8f d,
                            const float* bias, const float* rowscale) {
  const int lane = threadIdx.x & 31;
  const int n = lane & 15, mb = (lane & 16) ? 8 : 0;
  float b = bias[col0 + n];
  #pragma unroll
  for (int r = 0; r < 8; ++r)
    out[(mb + r) * ldo + col0 + n] = f2bf((d[r] + b) * rowscale[mb + r]);
}
DEV void storeD_bias_relu_bf(u16* out, int ldo, int col0, v8f d, const float* bias) {
  const int lane = threadIdx.x & 31;
  const int n = lane & 15, mb = (lane & 16) ? 8 : 0;
  float b = bias[col0 + n];
  #pragma unroll
  for (int r = 0; r < 8; ++r)
    out[(mb + r) * ldo + col0 + n] = f2bf(fmaxf(d[r] + b, 0.f));
}
DEV void storeD_out(float* og, int node0, int col0, v8f d, const float* bias) {
  const int lane = threadIdx.x & 31;
  const int n = lane & 15, mb = (lane & 16) ? 8 : 0;
  float b = bias[col0 + n];
  #pragma unroll
  for (int r = 0; r < 8; ++r)
    og[(size_t)(node0 + mb + r) * DN + col0 + n] = d[r] + b;
}

// ---------- prologue: weights -> bf16 in d_ws (Wk transposed), q_time const ----------
__global__ void tgat_prep(const float* __restrict__ Wq, const float* __restrict__ Wk,
                          const float* __restrict__ Wv, const float* __restrict__ Wo,
                          const float* __restrict__ W1, const float* __restrict__ W2,
                          const float* __restrict__ time_b, u16* __restrict__ ws) {
  int i = blockIdx.x * blockDim.x + threadIdx.x;
  if (i >= WS_ELEMS) return;
  float v;
  if (i < OWKT)       v = Wq[i];
  else if (i < OWV) { int e = i - OWKT; int j = e >> 8, d = e & 255; v = Wk[d * DKV + j]; }
  else if (i < OWO)   v = Wv[i - OWV];
  else if (i < OW1)   v = Wo[i - OWO];
  else if (i < OW2)   v = W1[i - OW1];
  else if (i < OQT)   v = W2[i - OW2];
  else                v = __builtin_cosf(time_b[i - OQT]);   // cos(0*w + b)
  ws[i] = f2bf(v);
}

// ---------- fused embedder: 16 nodes per block ----------
__global__ __launch_bounds__(256) void tgat_main(
    const int* __restrict__ idx, const float* __restrict__ t,
    const int* __restrict__ nbrs, const float* __restrict__ e_t,
    const int* __restrict__ e_id, const unsigned char* __restrict__ maskp,
    const float* __restrict__ node_feat, const float* __restrict__ memf,
    const float* __restrict__ event_feat,
    const float* __restrict__ time_w, const float* __restrict__ time_b,
    const float* __restrict__ bq, const float* __restrict__ bv,
    const float* __restrict__ bo, const float* __restrict__ b1,
    const float* __restrict__ b2,
    const u16* __restrict__ ws, float* __restrict__ out) {
  extern __shared__ char smem[];
  u16*   wt   = (u16*)  (smem + OFF_WT);    // 16x768 bf16 w~
  u16*   mo   = (u16*)  (smem + OFF_MO);    // alias of wt (post-attention)
  u16*   mi   = (u16*)  (smem + OFF_MI);    // alias
  u16*   h1   = (u16*)  (smem + OFF_H1);    // alias
  u16*   qc   = (u16*)  (smem + OFF_QC);
  u16*   qh   = (u16*)  (smem + OFF_QH);
  u16*   cb   = (u16*)  (smem + OFF_CB);
  int*   idxv = (int*)  (smem + OFF_IDX);
  int*   nbrL = (int*)  (smem + OFF_NBR);
  int*   eidL = (int*)  (smem + OFF_EID);
  float* dtL  = (float*)(smem + OFF_DT);
  float* mskf = (float*)(smem + OFF_MSK);
  float* valf = (float*)(smem + OFF_VAL);
  float* rowm = (float*)(smem + OFF_ROWM);

  const int tid  = threadIdx.x;
  const int lane = tid & 31;
  const int wid  = tid >> 5;            // wave32
  const int node0 = blockIdx.x * TILE;

  // ---- phase 0: per-tile metadata ----
  if (tid < TILE) idxv[tid] = idx[node0 + tid];
  for (int p = tid; p < TILE * KNB; p += 256) {
    int n = p / KNB, k = p % KNB;
    int g = (node0 + n) * KNB + k;
    nbrL[p] = nbrs[g];
    eidL[p] = e_id[g];
    dtL[p]  = t[node0 + n] - e_t[g];
    float mm = maskp[g] ? 1.f : 0.f;
    mskf[p] = mm;
    valf[p] = mm;
  }
  __syncthreads();
  if (tid < TILE) {
    float any = 0.f;
    for (int k = 0; k < KNB; ++k) any = fmaxf(any, mskf[tid * KNB + k]);
    rowm[tid] = any;                                  // 0 => no_nb => zero out row
    if (any == 0.f) for (int k = 0; k < KNB; ++k) valf[tid * KNB + k] = 1.f;
  }
  __syncthreads();

  // ---- phase 1: qc = [node_feat[idx]+memory[idx], cos(time_b)] (bf16) ----
  for (int row = wid; row < TILE; row += NWAVE) {
    size_t iv = (size_t)idxv[row] * DN;
    float4 a = *(const float4*)(node_feat + iv + lane * 4);
    float4 b = *(const float4*)(memf     + iv + lane * 4);
    u16* d0 = qc + row * DQ + lane * 4;
    d0[0] = f2bf(a.x + b.x); d0[1] = f2bf(a.y + b.y);
    d0[2] = f2bf(a.z + b.z); d0[3] = f2bf(a.w + b.w);
    const u16* qt = ws + OQT + lane * 4;
    u16* d1 = qc + row * DQ + DN + lane * 4;
    d1[0] = qt[0]; d1[1] = qt[1]; d1[2] = qt[2]; d1[3] = qt[3];
  }
  __syncthreads();

  // ---- G1: qh = qc @ Wq^T + bq   (16x256, K=256) ----
  for (int nt = wid; nt < 16; nt += NWAVE) {
    v8f acc = {};
    acc = gemm16(qc, DQ, ws + OWQ, DQ, nt * 16, 0, DQ, acc);
    storeD_bias_bf(qh, DQ, nt * 16, acc, bq);
  }
  __syncthreads();

  // ---- G2: w~_h = qh_h @ Wk_h   (16x768 bf16; B from pre-transposed WkT) ----
  for (int nt = wid; nt < 48; nt += NWAVE) {
    int h = nt / 24, j0 = (nt % 24) * 16;
    v8f acc = {};
    acc = gemm16(qh + h * 128, DQ, ws + OWKT, DQ, j0, h * 128, 128, acc);
    storeD_plain_bf(wt, 768, h * DKV + j0, acc);
  }
  __syncthreads();

  // ---- phase 4: attention (one wave per node; lane owns j = lane+32i of kvc) ----
  {
    float twr[4], tbr[4];
    #pragma unroll
    for (int i = 0; i < 4; ++i) { twr[i] = time_w[lane + 32 * i]; tbr[i] = time_b[lane + 32 * i]; }
    const float sc = 0.08838834764831845f;   // 1/sqrt(128)

    for (int n = wid; n < TILE; n += NWAVE) {
      float wj0[12], wj1[12];
      #pragma unroll
      for (int i = 0; i < 12; ++i) {
        int j = lane + 32 * i;
        wj0[i] = bf2f(wt[n * 768 + j]);
        wj1[i] = bf2f(wt[n * 768 + DKV + j]);
      }
      float my0 = -1e30f, my1 = -1e30f;
      for (int k = 0; k < KNB; ++k) {
        int   nb = nbrL[n * KNB + k], ei = eidL[n * KNB + k];
        float mk = mskf[n * KNB + k], dtv = dtL[n * KNB + k];
        const float* bn = node_feat  + (size_t)nb * DN;
        const float* bm = memf       + (size_t)nb * DN;
        const float* be = event_feat + (size_t)ei * DE;
        if (k + 1 < KNB) {                        // global_prefetch_b8 next rows
          int nb2 = nbrL[n * KNB + k + 1], ei2 = eidL[n * KNB + k + 1];
          __builtin_prefetch(node_feat  + (size_t)nb2 * DN + lane * 4, 0, 1);
          __builtin_prefetch(memf       + (size_t)nb2 * DN + lane * 4, 0, 1);
          __builtin_prefetch(event_feat + (size_t)ei2 * DE + lane * 4, 0, 1);
        }
        float a0 = 0.f, a1 = 0.f;
        #pragma unroll
        for (int i = 0; i < 12; ++i) {
          float v;
          if (i < 4)       v = (bn[lane + 32 * i] + bm[lane + 32 * i]) * mk;
          else if (i < 8)  v = __builtin_cosf(fmaf(dtv, twr[i - 4], tbr[i - 4]));
          else             v = be[lane + 32 * (i - 8)];
          a0 = fmaf(v, wj0[i], a0);
          a1 = fmaf(v, wj1[i], a1);
        }
        a0 = wsum(a0); a1 = wsum(a1);
        if (lane == k) { my0 = a0; my1 = a1; }
      }
      float vl = (lane < KNB) ? valf[n * KNB + lane] : 0.f;
      float l0 = (vl > 0.5f) ? my0 * sc : -1e30f;
      float l1 = (vl > 0.5f) ? my1 * sc : -1e30f;
      float m0 = wmaxr(l0), m1 = wmaxr(l1);
      float e0 = __expf(l0 - m0), e1 = __expf(l1 - m1);
      float s0 = wsum(e0), s1 = wsum(e1);
      float at0 = e0 / s0, at1 = e1 / s1;

      float c0[12], c1[12];
      #pragma unroll
      for (int i = 0; i < 12; ++i) { c0[i] = 0.f; c1[i] = 0.f; }
      for (int k = 0; k < KNB; ++k) {
        float p0 = __shfl(at0, k), p1 = __shfl(at1, k);
        int   nb = nbrL[n * KNB + k], ei = eidL[n * KNB + k];
        float mk = mskf[n * KNB + k], dtv = dtL[n * KNB + k];
        const float* bn = node_feat  + (size_t)nb * DN;
        const float* bm = memf       + (size_t)nb * DN;
        const float* be = event_feat + (size_t)ei * DE;
        #pragma unroll
        for (int i = 0; i < 12; ++i) {
          float v;
          if (i < 4)       v = (bn[lane + 32 * i] + bm[lane + 32 * i]) * mk;
          else if (i < 8)  v = __builtin_cosf(fmaf(dtv, twr[i - 4], tbr[i - 4]));
          else             v = be[lane + 32 * (i - 8)];
          c0[i] = fmaf(p0, v, c0[i]);
          c1[i] = fmaf(p1, v, c1[i]);
        }
      }
      #pragma unroll
      for (int i = 0; i < 12; ++i) {
        int j = lane + 32 * i;
        cb[n * 768 + j]       = f2bf(c0[i]);
        cb[n * 768 + DKV + j] = f2bf(c1[i]);
      }
    }
  }
  __syncthreads();   // w~ dead from here; mo/mi/h1 reuse its LDS

  // ---- G3: mha_h = c_h @ Wv_h^T + bv  (16x256, K=384) ----
  for (int nt = wid; nt < 16; nt += NWAVE) {
    int h = nt >> 3, o0 = (nt & 7) * 16;
    v8f acc = {};
    acc = gemm16(cb + h * DKV, 768, ws + OWV, DKV, h * 128 + o0, 0, DKV, acc);
    storeD_bias_bf(mo, DQ, nt * 16, acc, bv);
  }
  __syncthreads();

  // ---- G4: out = mha @ Wo^T + bo (zero no_nb rows) -> m_in[:,0:256]; concat q ----
  for (int nt = wid; nt < 16; nt += NWAVE) {
    v8f acc = {};
    acc = gemm16(mo, DQ, ws + OWO, DQ, nt * 16, 0, DQ, acc);
    storeD_bias_row_bf(mi, DKV, nt * 16, acc, bo, rowm);
  }
  for (int p = tid; p < TILE * DN; p += 256) {        // m_in[:,256:384] = q (bf16)
    int n = p >> 7, j = p & 127;
    mi[n * DKV + DQ + j] = qc[n * DQ + j];
  }
  __syncthreads();

  // ---- G5: h1 = relu(m_in @ W1^T + b1)  (16x128, K=384) ----
  for (int nt = wid; nt < 8; nt += NWAVE) {
    v8f acc = {};
    acc = gemm16(mi, DKV, ws + OW1, DKV, nt * 16, 0, DKV, acc);
    storeD_bias_relu_bf(h1, DN, nt * 16, acc, b1);
  }
  __syncthreads();

  // ---- G6: y = h1 @ W2^T + b2 -> global f32 ----
  for (int nt = wid; nt < 8; nt += NWAVE) {
    v8f acc = {};
    acc = gemm16(h1, DN, ws + OW2, DN, nt * 16, 0, DN, acc);
    storeD_out(out, node0, nt * 16, acc, b2);
  }
}

extern "C" void kernel_launch(void* const* d_in, const int* in_sizes, int n_in,
                              void* d_out, int out_size, void* d_ws, size_t ws_size,
                              hipStream_t stream) {
  const int*   idx        = (const int*)  d_in[0];
  const float* t          = (const float*)d_in[1];
  const int*   neighbors  = (const int*)  d_in[2];
  const float* e_t        = (const float*)d_in[3];
  const int*   e_id       = (const int*)  d_in[4];
  const unsigned char* mask = (const unsigned char*)d_in[5];   // bool
  const float* node_feat  = (const float*)d_in[6];
  const float* memory     = (const float*)d_in[7];
  const float* event_feat = (const float*)d_in[8];
  const float* time_w     = (const float*)d_in[9];
  const float* time_b     = (const float*)d_in[10];
  const float* Wq = (const float*)d_in[11]; const float* bq = (const float*)d_in[12];
  const float* Wk = (const float*)d_in[13]; /* bk: softmax-invariant, unused */
  const float* Wv = (const float*)d_in[15]; const float* bv = (const float*)d_in[16];
  const float* Wo = (const float*)d_in[17]; const float* bo = (const float*)d_in[18];
  const float* W1 = (const float*)d_in[19]; const float* b1 = (const float*)d_in[20];
  const float* W2 = (const float*)d_in[21]; const float* b2 = (const float*)d_in[22];
  u16* ws = (u16*)d_ws;                   // needs 786,688 B of scratch
  (void)in_sizes; (void)n_in; (void)out_size; (void)ws_size;

  // per-launch bf16 weight staging; ordered on the same stream
  tgat_prep<<<(WS_ELEMS + 255) / 256, 256, 0, stream>>>(Wq, Wk, Wv, Wo, W1, W2,
                                                        time_b, ws);

  (void)hipFuncSetAttribute((const void*)tgat_main,
                            hipFuncAttributeMaxDynamicSharedMemorySize, SMEM_BYTES);
  tgat_main<<<NQ / TILE, 256, SMEM_BYTES, stream>>>(
      idx, t, neighbors, e_t, e_id, mask, node_feat, memory, event_feat,
      time_w, time_b, bq, bv, bo, b1, b2, ws, (float*)d_out);
}
